// MambaActorCriticNetwork_29094108463117
// MI455X (gfx1250) — compile-verified, matrix operand order checked by reading, not run on hardware
//
#include <hip/hip_runtime.h>
#include <hip/hip_bf16.h>

// ---------------------------------------------------------------------------
// Mamba actor-critic forward for MI455X (gfx1250).
// GEMMs: v_wmma_f32_16x16x32_bf16 (wave32 WMMA), A staged via
// global_load_async_to_lds_b128 (ASYNCcnt), K-slab prefetch via
// global_prefetch_b8, all fragments loaded as ds_load_b128 pairs.
// ---------------------------------------------------------------------------

#define BB      8
#define LL      2048
#define IN_DIM  32
#define DD      512
#define EE      1024
#define NN      16
#define KK      4
#define RR      32
#define NLAYER  4
#define BLROWS  (BB * LL)      // 16384

typedef unsigned short u16;
typedef unsigned int   u32;
typedef unsigned long long u64;
typedef __attribute__((ext_vector_type(16))) __bf16 v16bf;
typedef __attribute__((ext_vector_type(8)))  float  v8f;

__device__ __forceinline__ u16 f2bf(float x) {
  u32 u = __float_as_uint(x);
  u32 r = u + 0x7FFFu + ((u >> 16) & 1u);   // round-to-nearest-even
  return (u16)(r >> 16);
}
__device__ __forceinline__ float silu_f(float x) { return x / (1.f + __expf(-x)); }

// ---------------------------- f32 -> bf16 ----------------------------------
__global__ void k_f32_to_bf16(const float* __restrict__ s, u16* __restrict__ d, int n) {
  int i = blockIdx.x * 256 + threadIdx.x;
  if (i < n) d[i] = f2bf(s[i]);
}

// ---------------------------- WMMA GEMM ------------------------------------
// C[M,N] f32 = A[M,K]bf16 (row major) x B[K,N]bf16 (row major) [+bias][+resid][softplus]
// Block: 256 thr (8 waves). Block tile 128 x (NTW*32). Wave tile 32 x (NTW*16).
// MODE: 0 = plain, 1 = residual add, 2 = softplus.  HB: bias present.
//
// A tile -> LDS via global_load_async_to_lds_b128 (per-lane LDS dest keeps the
// padded stride).  B is packed by consecutive-K pairs AND transposed per
// column: sBt[col][k/2] = B[k][col] | B[k+1][col]<<16, 16 kpairs contiguous.
// With the ISA 7.12.2 16-bit layouts, both A and B fragments then load as two
// ds_load_b128 each (kpair dwords {0..3}+4*kh and {8..11}+4*kh).

#define LDA32 20   // dword stride for A rows / B cols (80B: 16B aligned, conflict-free)

template <int NTW, int MODE, bool HB>
__global__ __launch_bounds__(256)
void k_gemm_bf16(const u16* __restrict__ A, const u16* __restrict__ Bw,
                 const float* __restrict__ bias, const float* __restrict__ resid,
                 float* __restrict__ C, int M, int N, int K)
{
  constexpr int BN    = NTW * 32;   // block N tile (128 or 64)
  constexpr int BHALF = BN / 2;     // column pairs per k-row
  constexpr int NBT   = BN / 32;    // B staging tasks per thread

  __shared__ u32 sA [128 * LDA32];
  __shared__ u32 sBt[BN  * LDA32];

  const int tid  = threadIdx.x;
  const int lane = tid & 31;
  const int w    = tid >> 5;
  const int wm   = w & 3;    // M subtile (x32 rows)
  const int wn   = w >> 2;   // N subtile (x NTW*16 cols)
  const int blockRow = blockIdx.y * 128;
  const int blockCol = blockIdx.x * BN;

  v8f acc[2][NTW];
#pragma unroll
  for (int i = 0; i < 2; ++i)
#pragma unroll
    for (int j = 0; j < NTW; ++j)
#pragma unroll
      for (int r = 0; r < 8; ++r) acc[i][j][r] = 0.f;

  const u32* Bg = (const u32*)Bw;  // bf16 pairs; N always even

  // ---- A staging (async to LDS): thread owns 2 B128 slots -----------------
  const int rowa = tid >> 2;       // 0..63
  const int grpa = tid & 3;        // 16B group within 32-element row slab
  const u32 sAbase = (u32)(uintptr_t)&sA[0];
  u32 lofA0 = sAbase + (u32)((rowa      ) * LDA32 + grpa * 4) * 4;
  u32 lofA1 = sAbase + (u32)((rowa + 64 ) * LDA32 + grpa * 4) * 4;
  u64 gaA0 = (u64)(uintptr_t)(A + (size_t)(blockRow + rowa     ) * K) + (u64)grpa * 16;
  u64 gaA1 = (u64)(uintptr_t)(A + (size_t)(blockRow + rowa + 64) * K) + (u64)grpa * 16;

  // ---- B staging: running pointers (advance 16*N dwords per K-step) -------
  const u32* pB0[NBT];
  const u32* pB1[NBT];
  u32 sOff[NBT];
#pragma unroll
  for (int i = 0; i < NBT; ++i) {
    int idx = tid + i * 256;
    int kp  = idx / BHALF;
    int cp  = idx % BHALF;
    pB0[i] = Bg + ((size_t)(2 * kp + 0) * N + blockCol) / 2 + cp;
    pB1[i] = Bg + ((size_t)(2 * kp + 1) * N + blockCol) / 2 + cp;
    sOff[i] = (u32)(2 * cp) * LDA32 + kp;       // transposed: [col][kpair]
  }

  for (int k0 = 0; k0 < K; k0 += 32) {
    // A tile -> LDS, asynchronously (bypasses VGPRs, tracked by ASYNCcnt).
    asm volatile("global_load_async_to_lds_b128 %0, %1, off"
                 :: "v"(lofA0), "v"(gaA0) : "memory");
    asm volatile("global_load_async_to_lds_b128 %0, %1, off"
                 :: "v"(lofA1), "v"(gaA1) : "memory");
    __builtin_prefetch((const void*)(uintptr_t)(gaA0 + 64), 0, 1);
    __builtin_prefetch((const void*)(uintptr_t)(gaA1 + 64), 0, 1);
    gaA0 += 64; gaA1 += 64;

    // B tile -> LDS, K-pair packed + per-column transposed.
#pragma unroll
    for (int i = 0; i < NBT; ++i) {
      u32 in0 = *pB0[i];
      u32 in1 = *pB1[i];
      sBt[sOff[i]        ] = (in0 & 0xFFFFu) | (in1 << 16);         // col 2cp
      sBt[sOff[i] + LDA32] = (in0 >> 16) | (in1 & 0xFFFF0000u);     // col 2cp+1
      __builtin_prefetch(pB0[i] + (size_t)16 * N, 0, 1);
      __builtin_prefetch(pB1[i] + (size_t)16 * N, 0, 1);
      pB0[i] += (size_t)16 * N;
      pB1[i] += (size_t)16 * N;
    }
    asm volatile("s_wait_asynccnt 0x0" ::: "memory");
    __syncthreads();

    // Fragments: 2x ds_load_b128 each (kpairs {0..3}+4kh, {8..11}+4kh).
    union Frag { v16bf v; u32 u[8]; } fa[2], fb[NTW];
    const int mlan = lane & 15, kh = lane >> 4;
#pragma unroll
    for (int mt = 0; mt < 2; ++mt) {
      const u32* ap = &sA[(wm * 32 + mt * 16 + mlan) * LDA32 + 4 * kh];
#pragma unroll
      for (int j2 = 0; j2 < 4; ++j2) { fa[mt].u[j2] = ap[j2]; fa[mt].u[j2 + 4] = ap[j2 + 8]; }
    }
#pragma unroll
    for (int nt = 0; nt < NTW; ++nt) {
      const u32* bp = &sBt[(wn * (NTW * 16) + nt * 16 + mlan) * LDA32 + 4 * kh];
#pragma unroll
      for (int j2 = 0; j2 < 4; ++j2) { fb[nt].u[j2] = bp[j2]; fb[nt].u[j2 + 4] = bp[j2 + 8]; }
    }
#pragma unroll
    for (int mt = 0; mt < 2; ++mt)
#pragma unroll
      for (int nt = 0; nt < NTW; ++nt)
        acc[mt][nt] = __builtin_amdgcn_wmma_f32_16x16x32_bf16(
            false, fa[mt].v, false, fb[nt].v, (short)0, acc[mt][nt], false, false);
    __syncthreads();
  }

  // Epilogue: C/D layout -> VGPR r holds M = r + 8*(lane>=16), N = lane&15.
  const int nlan = lane & 15, mh = lane >> 4;
#pragma unroll
  for (int nt = 0; nt < NTW; ++nt) {
    const u32 gcol = (u32)(blockCol + wn * (NTW * 16) + nt * 16 + nlan);
    float bv = 0.f;
    if constexpr (HB) bv = bias[gcol];
#pragma unroll
    for (int mt = 0; mt < 2; ++mt) {
      const u32 rbase = (u32)(blockRow + wm * 32 + mt * 16 + mh * 8);
#pragma unroll
      for (int r = 0; r < 8; ++r) {
        u32 offs = (rbase + (u32)r) * (u32)N + gcol;   // < 2^31: u32 math
        float v = acc[mt][nt][r] + bv;
        if constexpr (MODE == 1) v += resid[offs];
        if constexpr (MODE == 2) v = (v > 20.f) ? v : log1pf(__expf(v));
        C[offs] = v;
      }
    }
  }
}

// -------------------- causal depthwise conv (K=4) + SiLU -------------------
#define LCHUNK 128
__global__ __launch_bounds__(256)
void k_conv_silu(const float* __restrict__ xz, const float* __restrict__ cW,
                 const float* __restrict__ cb, float* __restrict__ U,
                 u16* __restrict__ Ubf)
{
  int e = blockIdx.x * 256 + threadIdx.x;
  int chunk = blockIdx.y % (LL / LCHUNK);
  int b = blockIdx.y / (LL / LCHUNK);
  int l0 = chunk * LCHUNK;
  float w0 = cW[e * 4 + 0], w1 = cW[e * 4 + 1], w2 = cW[e * 4 + 2], w3 = cW[e * 4 + 3];
  float bc = cb[e];
  const size_t base = (size_t)b * LL * 2 * EE + e;   // xs = xz[..., 0:E]
  float x3 = (l0 - 3 >= 0) ? xz[base + (size_t)(l0 - 3) * 2 * EE] : 0.f;
  float x2 = (l0 - 2 >= 0) ? xz[base + (size_t)(l0 - 2) * 2 * EE] : 0.f;
  float x1 = (l0 - 1 >= 0) ? xz[base + (size_t)(l0 - 1) * 2 * EE] : 0.f;
  for (int l = l0; l < l0 + LCHUNK; ++l) {
    float x0 = xz[base + (size_t)l * 2 * EE];
    float s = w0 * x3 + w1 * x2 + w2 * x1 + w3 * x0 + bc;
    float uv = silu_f(s);
    size_t o = (size_t)(b * LL + l) * EE + e;
    U[o] = uv;
    Ubf[o] = f2bf(uv);
    x3 = x2; x2 = x1; x1 = x0;
  }
}

// ------------------- split proj[BL,64] -> dt(bf16) / B / C -----------------
__global__ void k_split_proj(const float* __restrict__ P, u16* __restrict__ DT,
                             float* __restrict__ Bc, float* __restrict__ Cc)
{
  int i = blockIdx.x * 256 + threadIdx.x;   // over BLROWS*64 (exact multiple)
  int row = i >> 6, col = i & 63;
  float v = P[i];
  if (col < 32)      DT[row * 32 + col] = f2bf(v);
  else if (col < 48) Bc[row * 16 + col - 32] = v;
  else               Cc[row * 16 + col - 48] = v;
}

// ------------------- selective scan + output gate (fused) ------------------
#define TCH 64
__global__ __launch_bounds__(256)
void k_scan_gate(const float* __restrict__ U, const float* __restrict__ Dl,
                 const float* __restrict__ Bc, const float* __restrict__ Cc,
                 const float* __restrict__ Alog, const float* __restrict__ Dp,
                 const float* __restrict__ xz, u16* __restrict__ G)
{
  __shared__ float sB[TCH * NN];
  __shared__ float sC[TCH * NN];
  int e = blockIdx.x * 256 + threadIdx.x;
  int b = blockIdx.y;
  float A[NN], h[NN];
#pragma unroll
  for (int n = 0; n < NN; ++n) { A[n] = -__expf(Alog[e * NN + n]); h[n] = 0.f; }
  float dpe = Dp[e];
  const size_t zbase = (size_t)b * LL * 2 * EE + EE + e;
  for (int t0 = 0; t0 < LL; t0 += TCH) {
    __syncthreads();
#pragma unroll
    for (int i = 0; i < 4; ++i) {
      int idx = threadIdx.x + i * 256;    // 0..1023 = TCH*NN
      sB[idx] = Bc[((size_t)b * LL + t0) * NN + idx];
      sC[idx] = Cc[((size_t)b * LL + t0) * NN + idx];
    }
    __syncthreads();
    for (int tt = 0; tt < TCH; ++tt) {
      int t = t0 + tt;
      size_t o = ((size_t)b * LL + t) * EE + e;
      float d  = Dl[o];
      float ut = U[o];
      float du = d * ut;
      float y = 0.f;
#pragma unroll
      for (int n = 0; n < NN; ++n) {
        float dA = __expf(d * A[n]);
        h[n] = dA * h[n] + du * sB[tt * NN + n];
        y += h[n] * sC[tt * NN + n];
      }
      float z = xz[zbase + (size_t)t * 2 * EE];
      G[o] = f2bf((y + ut * dpe) * silu_f(z));
    }
  }
}

// --------------------- layernorm + actor head ------------------------------
__global__ __launch_bounds__(256)
void k_ln_actor(const float* __restrict__ X, const float* __restrict__ g,
                const float* __restrict__ bta, const float* __restrict__ aW,
                const float* __restrict__ ab, float* __restrict__ XN,
                float* __restrict__ out)
{
  __shared__ float red[256];
  int row = blockIdx.x;
  int t = threadIdx.x;
  const float* xr = X + (size_t)row * DD;
  float x0 = xr[t], x1 = xr[t + 256];
  red[t] = x0 + x1;
  __syncthreads();
  for (int s = 128; s > 0; s >>= 1) { if (t < s) red[t] += red[t + s]; __syncthreads(); }
  float mu = red[0] * (1.f / DD);
  __syncthreads();
  float d0 = x0 - mu, d1 = x1 - mu;
  red[t] = d0 * d0 + d1 * d1;
  __syncthreads();
  for (int s = 128; s > 0; s >>= 1) { if (t < s) red[t] += red[t + s]; __syncthreads(); }
  float rs = rsqrtf(red[0] * (1.f / DD) + 1e-5f);
  __syncthreads();
  float n0 = d0 * rs * g[t] + bta[t];
  float n1 = d1 * rs * g[t + 256] + bta[t + 256];
  XN[(size_t)row * DD + t] = n0;
  XN[(size_t)row * DD + t + 256] = n1;
  red[t] = n0 * aW[t] + n1 * aW[t + 256];
  __syncthreads();
  for (int s = 128; s > 0; s >>= 1) { if (t < s) red[t] += red[t + s]; __syncthreads(); }
  if (t == 0) out[row] = red[0] + ab[0];
}

// --------------------- mean over L (two-stage) -----------------------------
#define LPAR 16
__global__ void k_agg_part(const float* __restrict__ XN, float* __restrict__ PART)
{
  int i = blockIdx.x * 256 + threadIdx.x;   // BB*DD = 4096
  int c = blockIdx.y;                       // L chunk
  int b = i / DD, d = i % DD;
  float s = 0.f;
  int l0 = c * (LL / LPAR);
  for (int l = l0; l < l0 + LL / LPAR; ++l) s += XN[((size_t)b * LL + l) * DD + d];
  PART[(size_t)c * (BB * DD) + i] = s;
}
__global__ void k_agg_reduce(const float* __restrict__ PART, float* __restrict__ AGG)
{
  int i = blockIdx.x * 256 + threadIdx.x;   // BB*DD
  float s = 0.f;
#pragma unroll
  for (int c = 0; c < LPAR; ++c) s += PART[(size_t)c * (BB * DD) + i];
  AGG[i] = s * (1.f / LL);
}

// --------------------- critic MLP ------------------------------------------
__global__ __launch_bounds__(256)
void k_critic(const float* __restrict__ AGG, const float* __restrict__ W1,
              const float* __restrict__ b1, const float* __restrict__ W2,
              const float* __restrict__ b2, float* __restrict__ out)
{
  __shared__ float red[256];
  int b = blockIdx.x, j = threadIdx.x;
  float s = b1[j];
  for (int i = 0; i < DD; ++i) s += AGG[b * DD + i] * W1[i * 256 + j];
  float h = fmaxf(s, 0.f);
  red[j] = h * W2[j];
  __syncthreads();
  for (int st = 128; st > 0; st >>= 1) { if (j < st) red[j] += red[j + st]; __syncthreads(); }
  if (j == 0) out[b] = red[0] + b2[0];
}

// ===========================================================================
extern "C" void kernel_launch(void* const* d_in, const int* in_sizes, int n_in,
                              void* d_out, int out_size, void* d_ws, size_t ws_size,
                              hipStream_t stream)
{
  (void)in_sizes; (void)n_in; (void)out_size; (void)ws_size;

  const float* state    = (const float*)d_in[0];
  const float* in_W     = (const float*)d_in[1];
  const float* in_b     = (const float*)d_in[2];
  const float* m_in_W   = (const float*)d_in[3];
  const float* m_in_b   = (const float*)d_in[4];
  const float* m_conv_W = (const float*)d_in[5];
  const float* m_conv_b = (const float*)d_in[6];
  const float* m_x_W    = (const float*)d_in[7];
  const float* m_dt_W   = (const float*)d_in[8];
  const float* m_dt_b   = (const float*)d_in[9];
  const float* m_Alog   = (const float*)d_in[10];
  const float* m_D      = (const float*)d_in[11];
  const float* m_out_W  = (const float*)d_in[12];
  const float* m_out_b  = (const float*)d_in[13];
  const float* ln_g     = (const float*)d_in[14];
  const float* ln_b     = (const float*)d_in[15];
  const float* actor_W  = (const float*)d_in[16];
  const float* actor_b  = (const float*)d_in[17];
  const float* c1_W     = (const float*)d_in[18];
  const float* c1_b     = (const float*)d_in[19];
  const float* c2_W     = (const float*)d_in[20];
  const float* c2_b     = (const float*)d_in[21];
  float* out = (float*)d_out;

  char* ws = (char*)d_ws;
  size_t off = 0;
  auto alloc = [&](size_t bytes) -> char* {
    char* p = ws + off;
    off += (bytes + 255) & ~(size_t)255;
    return p;
  };
  float* X     = (float*)alloc((size_t)BLROWS * DD * 4);
  float* XZ    = (float*)alloc((size_t)BLROWS * 2 * EE * 4);
  float* U     = (float*)alloc((size_t)BLROWS * EE * 4);
  u16*   Ubf   = (u16*)  alloc((size_t)BLROWS * EE * 2);
  float* PROJ  = (float*)alloc((size_t)BLROWS * 64 * 4);
  u16*   DTbf  = (u16*)  alloc((size_t)BLROWS * RR * 2);
  float* Bc    = (float*)alloc((size_t)BLROWS * NN * 4);
  float* Cc    = (float*)alloc((size_t)BLROWS * NN * 4);
  float* DELTA = (float*)alloc((size_t)BLROWS * EE * 4);
  u16*   Gbf   = (u16*)  alloc((size_t)BLROWS * EE * 2);
  float* XN    = (float*)alloc((size_t)BLROWS * DD * 4);
  float* PART  = (float*)alloc((size_t)LPAR * BB * DD * 4);
  float* AGG   = (float*)alloc((size_t)BB * DD * 4);
  u16*   Xbf   = (u16*)  alloc((size_t)BLROWS * DD * 2);
  u16*   Sbf   = (u16*)  alloc((size_t)BLROWS * IN_DIM * 2);
  u16*   inWbf = (u16*)  alloc((size_t)IN_DIM * DD * 2);
  u16*   minWbf  = (u16*)alloc((size_t)NLAYER * DD * 2 * EE * 2);
  u16*   mxWbf   = (u16*)alloc((size_t)NLAYER * EE * 64 * 2);
  u16*   mdtWbf  = (u16*)alloc((size_t)NLAYER * RR * EE * 2);
  u16*   moutWbf = (u16*)alloc((size_t)NLAYER * EE * DD * 2);

  auto cvt = [&](const float* s, u16* d, int n) {
    k_f32_to_bf16<<<(n + 255) / 256, 256, 0, stream>>>(s, d, n);
  };
  cvt(state,   Sbf,     BLROWS * IN_DIM);
  cvt(in_W,    inWbf,   IN_DIM * DD);
  cvt(m_in_W,  minWbf,  NLAYER * DD * 2 * EE);
  cvt(m_x_W,   mxWbf,   NLAYER * EE * 64);
  cvt(m_dt_W,  mdtWbf,  NLAYER * RR * EE);
  cvt(m_out_W, moutWbf, NLAYER * EE * DD);

  const dim3 gN512 (DD / 128,      BLROWS / 128);
  const dim3 gN2048(2 * EE / 128,  BLROWS / 128);
  const dim3 gN1024(EE / 128,      BLROWS / 128);
  const dim3 gN64  (64 / 64,       BLROWS / 128);

  // Input projection: [BL,32] x [32,512]
  k_gemm_bf16<4, 0, true><<<gN512, 256, 0, stream>>>(
      Sbf, inWbf, in_b, nullptr, X, BLROWS, DD, IN_DIM);

  for (int l = 0; l < NLAYER; ++l) {
    cvt(X, Xbf, BLROWS * DD);
    // xz = x @ in_W + b : [BL,512] x [512,2048]
    k_gemm_bf16<4, 0, true><<<gN2048, 256, 0, stream>>>(
        Xbf, minWbf + (size_t)l * DD * 2 * EE, m_in_b + l * 2 * EE, nullptr,
        XZ, BLROWS, 2 * EE, DD);
    // causal depthwise conv + silu -> u
    dim3 gc(EE / 256, BB * (LL / LCHUNK));
    k_conv_silu<<<gc, 256, 0, stream>>>(XZ, m_conv_W + (size_t)l * EE * KK,
                                        m_conv_b + l * EE, U, Ubf);
    // proj = u @ x_W : [BL,1024] x [1024,64] (no bias)
    k_gemm_bf16<2, 0, false><<<gN64, 256, 0, stream>>>(
        Ubf, mxWbf + (size_t)l * EE * 64, nullptr, nullptr, PROJ, BLROWS, 64, EE);
    k_split_proj<<<(BLROWS * 64) / 256, 256, 0, stream>>>(PROJ, DTbf, Bc, Cc);
    // delta = softplus(dt @ dt_W + b) : [BL,32] x [32,1024]
    k_gemm_bf16<4, 2, true><<<gN1024, 256, 0, stream>>>(
        DTbf, mdtWbf + (size_t)l * RR * EE, m_dt_b + l * EE, nullptr,
        DELTA, BLROWS, EE, RR);
    // selective scan + gate fused: emits gated bf16 activation
    dim3 gs(EE / 256, BB);
    k_scan_gate<<<gs, 256, 0, stream>>>(U, DELTA, Bc, Cc,
                                        m_Alog + (size_t)l * EE * NN, m_D + l * EE,
                                        XZ, Gbf);
    // x = x + gated @ out_W + b : [BL,1024] x [1024,512], residual fused
    k_gemm_bf16<4, 1, true><<<gN512, 256, 0, stream>>>(
        Gbf, moutWbf + (size_t)l * EE * DD, m_out_b + l * DD, X, X, BLROWS, DD, EE);
  }

  k_ln_actor<<<BLROWS, 256, 0, stream>>>(X, ln_g, ln_b, actor_W, actor_b, XN, out);
  dim3 ga((BB * DD) / 256, LPAR);
  k_agg_part<<<ga, 256, 0, stream>>>(XN, PART);
  k_agg_reduce<<<(BB * DD) / 256, 256, 0, stream>>>(PART, AGG);
  k_critic<<<BB, 256, 0, stream>>>(AGG, c1_W, c1_b, c2_W, c2_b, out + BLROWS);
}
